// PNB_66803921322210
// MI455X (gfx1250) — matrix operand order repaired and checked by reading, not compiled, not run on hardware
//
#include <hip/hip_runtime.h>
#include <hip/hip_bf16.h>

typedef __attribute__((ext_vector_type(16))) _Float16 v16h;
typedef __attribute__((ext_vector_type(8)))  _Float16 v8h;   // 16 bytes, 16B aligned
typedef __attribute__((ext_vector_type(8)))  float    v8f;

#define NB   4
#define NC   64
#define NF   32
#define NPIX 16384   // 128*128

__device__ __forceinline__ v8f wmma16(v16h a, v16h b, v8f c) {
  return __builtin_amdgcn_wmma_f32_16x16x32_f16(false, a, false, b, (short)0, c,
                                                false, false);
}

// ---- vectorized fragment loaders (CDNA5 16-bit WMMA layouts) ----------------
// A: 16x32 (MxK), source row-major [m][k].  Per lane: two contiguous 8-half
// (16B) runs at k = 8*hf and k = 16 + 8*hf.  ld must be a multiple of 8.
__device__ __forceinline__ v16h frag_a(const _Float16* s, int ld) {
  int lane = threadIdx.x & 31;
  int m = lane & 15, hf = lane >> 4;
  const _Float16* r = s + m * ld + (hf << 3);
  v8h lo = *(const v8h*)r;
  v8h hi = *(const v8h*)(r + 16);
  v16h a;
#pragma unroll
  for (int e = 0; e < 8; ++e) { a[e] = lo[e]; a[e + 8] = hi[e]; }
  return a;
}
// B: 32x16 (KxN) gathered from reduction-major storage [n][k].  Per lane: two
// contiguous 8-half (16B) runs at k = 8*hf and k = 16 + 8*hf.  ld must be a multiple of 8.
__device__ __forceinline__ v16h frag_bT(const _Float16* s, int ld) {
  int lane = threadIdx.x & 31;
  int n = lane & 15, hf = lane >> 4;
  const _Float16* r = s + n * ld + (hf << 3);
  v8h lo = *(const v8h*)r;
  v8h hi = *(const v8h*)(r + 16);
  v16h b;
#pragma unroll
  for (int e = 0; e < 8; ++e) { b[e] = lo[e]; b[e + 8] = hi[e]; }
  return b;
}

// =============================================================================
// Kernel 1: theta = 1x1 conv -> theta f16 [B][N][C]
// block = 256 (8 waves), 128 pixels per block; grid = 512
// =============================================================================
__global__ void theta_conv_kernel(const float* __restrict__ x,
                                  const float* __restrict__ w1,
                                  const float* __restrict__ b1,
                                  _Float16* __restrict__ theta) {
  __shared__ __align__(16) _Float16 sX[128][72];  // [pixel][c]
  __shared__ __align__(16) _Float16 sW[64][72];   // [oc][c] (w1's native layout)
  int tid = threadIdx.x;
  int b  = blockIdx.x >> 7;
  int n0 = (blockIdx.x & 127) << 7;

  for (int idx = tid; idx < 128 * 64; idx += 256) {
    int p = idx & 127, c = idx >> 7;
    sX[p][c] = (_Float16)x[((size_t)(b * NC + c)) * NPIX + n0 + p];
  }
  for (int idx = tid; idx < 64 * 64; idx += 256) {
    int oc = idx >> 6, c = idx & 63;
    sW[oc][c] = (_Float16)w1[idx];
  }
  __syncthreads();

  int wave = tid >> 5, lane = tid & 31;
  int m0 = wave << 4;
  int n = lane & 15, hf = lane >> 4;
#pragma unroll
  for (int nt = 0; nt < 4; ++nt) {
    v8f acc = {};
#pragma unroll
    for (int kt = 0; kt < 2; ++kt)
      acc = wmma16(frag_a(&sX[m0][kt * 32], 72), frag_bT(&sW[nt * 16][kt * 32], 72), acc);
    float bb = b1[nt * 16 + n];
#pragma unroll
    for (int r = 0; r < 8; ++r) {
      int row = r + (hf << 3);
      theta[((size_t)(b * NPIX + n0 + m0 + row)) * NC + nt * 16 + n] =
          (_Float16)(acc[r] + bb);
    }
  }
}

// =============================================================================
// Kernel 2: patch conv (kernel==stride==S) -> key f16 [B][M][C], valT [B][F][M]
// GEMM over K = C*S*S.  block = 192 (6 waves: 4 key col-tiles + 2 val col-tiles)
// grid = B * M/16
// =============================================================================
template <int S>
__global__ void kv_conv_kernel(const float* __restrict__ x,
                               const float* __restrict__ wk, const float* __restrict__ bk,
                               const float* __restrict__ wg, const float* __restrict__ bg,
                               _Float16* __restrict__ key, _Float16* __restrict__ valT) {
  constexpr int K    = NC * S * S;
  constexpr int Wm   = 128 / S;
  constexpr int Mtot = Wm * Wm;
  __shared__ __align__(16) _Float16 sA[16][40];    // 16 patches x 32-K chunk
  __shared__ __align__(16) _Float16 sWc[96][40];   // rows 0-63: wk, 64-95: wg

  int b  = blockIdx.x / (Mtot / 16);
  int m0 = (blockIdx.x % (Mtot / 16)) << 4;
  int wave = threadIdx.x >> 5, lane = threadIdx.x & 31;
  bool iskey = (wave < 4);
  int oc0 = iskey ? (wave << 4) : ((wave - 4) << 4);
  int wrow = iskey ? oc0 : (64 + oc0);

  v8f acc = {};
  for (int kc = 0; kc < K; kc += 32) {
    __syncthreads();
    for (int idx = threadIdx.x; idx < 512; idx += blockDim.x) {
      int row = idx >> 5, kk = idx & 31;
      int k = kc + kk;
      int c = k / (S * S), rem = k % (S * S);
      int ki = rem / S, kj = rem % S;
      int m = m0 + row, py = m / Wm, px = m % Wm;
      sA[row][kk] =
          (_Float16)x[(((size_t)(b * NC + c)) * 128 + (py * S + ki)) * 128 + px * S + kj];
    }
    for (int idx = threadIdx.x; idx < 96 * 32; idx += blockDim.x) {
      int row = idx >> 5, kk = idx & 31;
      sWc[row][kk] = (_Float16)((row < 64) ? wk[(size_t)row * K + kc + kk]
                                           : wg[(size_t)(row - 64) * K + kc + kk]);
    }
    __syncthreads();
    acc = wmma16(frag_a(&sA[0][0], 40), frag_bT(&sWc[wrow][0], 40), acc);
  }

  int n = lane & 15, hf = lane >> 4;
  float bb = (iskey ? bk : bg)[oc0 + n];
#pragma unroll
  for (int r = 0; r < 8; ++r) {
    int m = m0 + r + (hf << 3);
    if (iskey)
      key[((size_t)(b * Mtot + m)) * NC + oc0 + n] = (_Float16)(acc[r] + bb);
    else
      valT[((size_t)(b * NF + oc0 + n)) * Mtot + m] = (_Float16)(acc[r] + bb);
  }
}

// =============================================================================
// Kernel 3: flash attention with TRANSPOSED logits.
//   Lt[m][n] = K[m][c] . theta[n][c]   (A = key rows, B = theta^T)
//   softmax over m  ==  in-lane reduction over 16 VGPR values + one shfl_xor(16)
//   P lands reduction-major -> 2x b128 LDS stores, B-frag = 2x b128 loads
//   O^T[f][n] += V^T[f][m] . P[m][n]   (A = valT rows, reduction-major)
// one wave = 16 query pixels; block = 256 (8 waves), grid = 512
// =============================================================================
__global__ void attn_kernel(const _Float16* __restrict__ theta,
                            const _Float16* __restrict__ key,
                            const _Float16* __restrict__ valT,
                            _Float16* __restrict__ fusion, int M, int foff) {
  __shared__ __align__(16) _Float16 sP[8][16][40];   // per-wave P tile [n][m']
  int wave = threadIdx.x >> 5, lane = threadIdx.x & 31;
  int tile = (blockIdx.x << 3) + wave;
  int b  = tile >> 10;                  // N/16 = 1024 tiles per batch
  int n0 = (tile & 1023) << 4;
  int n = lane & 15, hf = lane >> 4;

  const _Float16* thb = theta + ((size_t)(b * NPIX + n0)) * NC;
  const _Float16* kb  = key + (size_t)b * M * NC;
  const _Float16* vb  = valT + (size_t)b * NF * M;

  // query fragments (B operands), loaded once
  v16h bq0 = frag_bT(thb, NC);        // c = 0..31
  v16h bq1 = frag_bT(thb + 32, NC);   // c = 32..63

  v8f o0 = {}, o1 = {};               // O^T rows f=0..15 / f=16..31
  float run_m = -3.0e38f, run_s = 0.0f;  // per query column (this lane's n)

  for (int mt = 0; mt < M; mt += 32) {
    if (mt + 32 < M) __builtin_prefetch(kb + (size_t)(mt + 32) * NC, 0, 1);
    // Lt tiles: l0 = keys mt..mt+15, l1 = keys mt+16..mt+31 (C-rows = keys)
    v8f l0 = {}, l1 = {};
    l0 = wmma16(frag_a(kb + (size_t)mt * NC, NC), bq0, l0);
    l0 = wmma16(frag_a(kb + (size_t)mt * NC + 32, NC), bq1, l0);
    l1 = wmma16(frag_a(kb + (size_t)(mt + 16) * NC, NC), bq0, l1);
    l1 = wmma16(frag_a(kb + (size_t)(mt + 16) * NC + 32, NC), bq1, l1);

    // online softmax over this tile's 32 keys for query column n
    float mx = fmaxf(l0[0], l1[0]);
#pragma unroll
    for (int r = 1; r < 8; ++r) mx = fmaxf(mx, fmaxf(l0[r], l1[r]));
    mx = fmaxf(mx, __shfl_xor(mx, 16, 32));
    float nm = fmaxf(run_m, mx);
    float corr = __expf(run_m - nm);
    v8h p0, p1;
    float s = 0.0f;
#pragma unroll
    for (int r = 0; r < 8; ++r) {
      float e0 = __expf(l0[r] - nm);   // key m' = 8*hf + r
      float e1 = __expf(l1[r] - nm);   // key m' = 16 + 8*hf + r
      s += e0 + e1;
      p0[r] = (_Float16)e0;
      p1[r] = (_Float16)e1;
    }
    s += __shfl_xor(s, 16, 32);
    run_s = run_s * corr + s;
    run_m = nm;
#pragma unroll
    for (int r = 0; r < 8; ++r) { o0[r] *= corr; o1[r] *= corr; }

    // stage P reduction-major: sP[n][m'] — two contiguous 16B stores per lane
    *(v8h*)&sP[wave][n][(hf << 3)]      = p0;
    *(v8h*)&sP[wave][n][16 + (hf << 3)] = p1;

    // O^T += V^T x P   (B-frag: 2x b128 loads; in-wave DS ordering suffices)
    v16h bp = frag_bT(&sP[wave][0][0], 40);
    o0 = wmma16(frag_a(vb + mt, M), bp, o0);
    o1 = wmma16(frag_a(vb + (size_t)16 * M + mt, M), bp, o1);
  }

  // normalize and store: O^T col n is this lane's pixel; f-runs are contiguous
  float inv = 1.0f / run_s;
  v8h r0, r1;
#pragma unroll
  for (int r = 0; r < 8; ++r) {
    r0[r] = (_Float16)(o0[r] * inv);   // f = 8*hf + r
    r1[r] = (_Float16)(o1[r] * inv);   // f = 16 + 8*hf + r
  }
  _Float16* fb = fusion + ((size_t)(b * NPIX + n0 + n)) * 96 + foff;
  *(v8h*)(fb + (hf << 3))      = r0;
  *(v8h*)(fb + 16 + (hf << 3)) = r1;
}

// =============================================================================
// Kernel 4: out = alpha * (wo . fusion + bo) + input.  K=96, cols=C=64.
// block = 256 (8 waves), 128 pixels per block; grid = 512
// =============================================================================
__global__ void out_conv_kernel(const _Float16* __restrict__ fusion,
                                const float* __restrict__ wo,
                                const float* __restrict__ bo,
                                const float* __restrict__ x,
                                const float* __restrict__ alpha,
                                float* __restrict__ out) {
  __shared__ __align__(16) _Float16 sW[64][104];  // [oc][f] (wo's native layout)
  int tid = threadIdx.x;
  int b  = blockIdx.x >> 7;
  int n0 = (blockIdx.x & 127) << 7;

  for (int idx = tid; idx < 64 * 96; idx += 256) {
    int oc = idx / 96, f = idx - oc * 96;
    sW[oc][f] = (_Float16)wo[idx];
  }
  __syncthreads();

  int wave = tid >> 5, lane = tid & 31;
  int m0 = wave << 4;
  int n = lane & 15, hf = lane >> 4;
  float al = alpha[0];

  const _Float16* ab = fusion + ((size_t)(b * NPIX + n0 + m0)) * 96;
  v16h a0 = frag_a(ab, 96), a1 = frag_a(ab + 32, 96), a2 = frag_a(ab + 64, 96);

#pragma unroll
  for (int nt = 0; nt < 4; ++nt) {
    v8f acc = {};
    acc = wmma16(a0, frag_bT(&sW[nt * 16][0], 104), acc);
    acc = wmma16(a1, frag_bT(&sW[nt * 16][32], 104), acc);
    acc = wmma16(a2, frag_bT(&sW[nt * 16][64], 104), acc);
    float bb = bo[nt * 16 + n];
#pragma unroll
    for (int r = 0; r < 8; ++r) {
      int pix = n0 + m0 + r + (hf << 3);
      int oc = nt * 16 + n;
      size_t o = ((size_t)(b * NC + oc)) * NPIX + pix;
      out[o] = al * (acc[r] + bb) + x[o];
    }
  }
}

// =============================================================================
extern "C" void kernel_launch(void* const* d_in, const int* in_sizes, int n_in,
                              void* d_out, int out_size, void* d_ws, size_t ws_size,
                              hipStream_t stream) {
  const float* x  = (const float*)d_in[0];
  const float* w1 = (const float*)d_in[1];  const float* b1 = (const float*)d_in[2];
  const float* w2 = (const float*)d_in[3];  const float* b2 = (const float*)d_in[4];
  const float* w3 = (const float*)d_in[5];  const float* b3 = (const float*)d_in[6];
  const float* w4 = (const float*)d_in[7];  const float* b4 = (const float*)d_in[8];
  const float* w5 = (const float*)d_in[9];  const float* b5 = (const float*)d_in[10];
  const float* w6 = (const float*)d_in[11]; const float* b6 = (const float*)d_in[12];
  const float* w7 = (const float*)d_in[13]; const float* b7 = (const float*)d_in[14];
  const float* wo = (const float*)d_in[15]; const float* bo = (const float*)d_in[16];
  const float* alpha = (const float*)d_in[17];
  float* out = (float*)d_out;

  char* p = (char*)d_ws;
  _Float16* theta  = (_Float16*)p; p += (size_t)NB * NPIX * NC * 2;   // 8 MB
  _Float16* fusion = (_Float16*)p; p += (size_t)NB * NPIX * 96 * 2;   // 12 MB
  _Float16* key4   = (_Float16*)p; p += (size_t)NB * 1024 * NC * 2;
  _Float16* val4   = (_Float16*)p; p += (size_t)NB * 1024 * NF * 2;
  _Float16* key8   = (_Float16*)p; p += (size_t)NB * 256 * NC * 2;
  _Float16* val8   = (_Float16*)p; p += (size_t)NB * 256 * NF * 2;
  _Float16* key16  = (_Float16*)p; p += (size_t)NB * 64 * NC * 2;
  _Float16* val16  = (_Float16*)p; p += (size_t)NB * 64 * NF * 2;

  theta_conv_kernel<<<512, 256, 0, stream>>>(x, w1, b1, theta);
  kv_conv_kernel<4><<<NB * 1024 / 16, 192, 0, stream>>>(x, w2, b2, w3, b3, key4, val4);
  kv_conv_kernel<8><<<NB * 256 / 16, 192, 0, stream>>>(x, w4, b4, w5, b5, key8, val8);
  kv_conv_kernel<16><<<NB * 64 / 16, 192, 0, stream>>>(x, w6, b6, w7, b7, key16, val16);
  attn_kernel<<<512, 256, 0, stream>>>(theta, key4, val4, fusion, 1024, 0);
  attn_kernel<<<512, 256, 0, stream>>>(theta, key8, val8, fusion, 256, 32);
  attn_kernel<<<512, 256, 0, stream>>>(theta, key16, val16, fusion, 64, 64);
  out_conv_kernel<<<512, 256, 0, stream>>>(fusion, wo, bo, x, alpha, out);
}
